// Informer_44813688766539
// MI455X (gfx1250) — matrix operand order repaired and checked by reading, not compiled, not run on hardware
//
#include <hip/hip_runtime.h>
#include <math.h>

typedef _Float16 half_t;
typedef _Float16 h8  __attribute__((ext_vector_type(8)));
typedef _Float16 v16h __attribute__((ext_vector_type(16)));
typedef float    v8f  __attribute__((ext_vector_type(8)));
typedef unsigned int u32x4 __attribute__((ext_vector_type(4)));
typedef unsigned int u32x8 __attribute__((ext_vector_type(8)));

constexpr int Bn = 4, Ln = 8192, Dm = 64, Hn = 8, NLn = 2, DFFn = 256, NCn = 5, En = 8, Un = 50;
constexpr int Mrows = Bn * Ln;                 // 32768

// ---------------------------------------------------------------------------
// Tensor Data Mover: issue a 2-D tile load (global -> LDS) via D# descriptor.
// data_size = 2 bytes (f16). Row padding in LDS via pad_interval/pad_amount.
//   pad_int enc: pads after 2^(enc+1) DWORDs;  pad_amt enc: (enc+1) DWORDs pad.
// Tracked on TENSORcnt; caller pairs with s_wait_tensorcnt + barrier.
// ---------------------------------------------------------------------------
__device__ __forceinline__ void tdm_load_tile_f16(
    unsigned lds_off, const void* gaddr,
    unsigned tile_d0, unsigned tile_d1, unsigned stride_elems,
    unsigned pad_int, unsigned pad_amt)
{
    unsigned long long ga = (unsigned long long)gaddr;
    u32x4 g0;
    g0[0] = 1u;                                     // count=1 (valid user descriptor)
    g0[1] = lds_off;                                // lds_addr (bytes, wave-relative)
    g0[2] = (unsigned)ga;                           // global_addr[31:0]   (bits 95:64)
    g0[3] = ((unsigned)(ga >> 32) & 0x01FFFFFFu)    // global_addr[56:32]
            | 0x80000000u;                          // type=2 ("image")    (bits 127:126)
    const unsigned td0 = stride_elems;              // tensor_dim0 >= tile cols
    const unsigned td1 = 1u << 20;                  // tensor_dim1: ample (tiles in-bounds)
    u32x8 g1;
    g1[0] = (1u << 16)                              // data_size = 2 bytes
          | (1u << 20)                              // pad_enable
          | (pad_int << 22) | (pad_amt << 25);
    g1[1] = (td0 & 0xFFFFu) << 16;                  // tensor_dim0[15:0]  (atomic bar addr = 0)
    g1[2] = ((td0 >> 16) & 0xFFFFu) | ((td1 & 0xFFFFu) << 16);
    g1[3] = ((td1 >> 16) & 0xFFFFu) | (tile_d0 << 16);   // tile_dim0
    g1[4] = tile_d1;                                // tile_dim1 (tile_dim2 = 0)
    g1[5] = stride_elems;                           // tensor_dim0_stride[31:0]
    g1[6] = 0u;
    g1[7] = 0u;
    asm volatile("tensor_load_to_lds %0, %1" :: "s"(g0), "s"(g1) : "memory");
}

// ---------------------------------------------------------------------------
// Embedding: circular conv1d (width 3) + sinusoidal positional encoding
// ---------------------------------------------------------------------------
__global__ __launch_bounds__(256) void embed_kernel(
    const float* __restrict__ x_enc, const float* __restrict__ emb_w,
    float* __restrict__ xf32, half_t* __restrict__ xf16)
{
    int t = blockIdx.x * 256 + threadIdx.x;
    if (t >= Mrows * Dm) return;
    int d   = t & 63;
    int row = t >> 6;
    int l   = row & (Ln - 1);
    int b   = row >> 13;                        // L = 2^13
    const float* xb = x_enc + (size_t)b * Ln;
    int lm1 = (l - 1 + Ln) & (Ln - 1);
    int lp1 = (l + 1) & (Ln - 1);
    float v = emb_w[d * 3 + 0] * xb[lm1] + emb_w[d * 3 + 1] * xb[l] + emb_w[d * 3 + 2] * xb[lp1];
    int i2 = d & ~1;
    float div = __expf((float)i2 * -0.14391157f);   // -ln(10000)/64
    float ang = (float)l * div;
    float pe  = (d & 1) ? __cosf(ang) : __sinf(ang);
    float o = v + pe;
    xf32[t] = o;
    xf16[t] = (half_t)o;
}

// ---------------------------------------------------------------------------
// Weight pre-pack: fp32 (K x N) -> f16 transposed (N x K)
// ---------------------------------------------------------------------------
__global__ void pack_wT_kernel(const float* __restrict__ W, half_t* __restrict__ out, int K, int N)
{
    int t = blockIdx.x * 256 + threadIdx.x;
    if (t >= K * N) return;
    int n = t / K, k = t % K;
    out[t] = (half_t)W[(size_t)k * N + n];
}

__global__ void copy_f32_kernel(const float* __restrict__ s, float* __restrict__ d, int n)
{
    int t = blockIdx.x * 256 + threadIdx.x;
    if (t < n) d[t] = s[t];
}

// ---------------------------------------------------------------------------
// WMMA GEMM: C[M x N] = A[M x K](f16) * Bt[N x K](f16)^T + bias
// block = 256 threads (8 waves), tile = 128 x 64, K stepped by 32.
// Staging via Tensor Data Mover (double-buffered A, one-shot B slice).
// flags: 1 = write fp32, 2 = write f16, 4 = relu
// ---------------------------------------------------------------------------
__global__ __launch_bounds__(256) void gemm_wmma_kernel(
    const half_t* __restrict__ A, const half_t* __restrict__ Bt,
    const float* __restrict__ bias,
    float* __restrict__ Cf, half_t* __restrict__ Ch,
    int Mtot, int Ntot, int K, int flags)
{
    __shared__ half_t As[2][128 * 56];          // 128 rows x 32 K-halves, pitch 56 (112B)
    __shared__ half_t Bs[64 * 264];             // 64 rows x K halves, pitch K+8

    const int bpitch = K + 8;
    const int tid  = threadIdx.x;
    const int wave = tid >> 5;
    const int lane = tid & 31;
    const int m0 = blockIdx.x * 128;
    const int n0 = blockIdx.y * 64;
    const int S  = K >> 5;                      // K-steps of 32

    if (wave == 0) {
        // B slice: 64 rows x K cols, LDS row pitch K+8 halves (pad 16B per row).
        //   K=64: rows 128B -> pad_int enc 4 (32 DW); K=256: rows 512B -> enc 6 (128 DW).
        tdm_load_tile_f16((unsigned)(size_t)&Bs[0],
                          (const void*)(Bt + (size_t)n0 * K),
                          (unsigned)K, 64u, (unsigned)K,
                          (K == 64) ? 4u : 6u, 3u);
        // A tile 0: 128 rows x 32 cols, LDS pitch 56 halves (64B data + 48B pad).
        tdm_load_tile_f16((unsigned)(size_t)&As[0][0],
                          (const void*)(A + (size_t)m0 * K),
                          32u, 128u, (unsigned)K, 3u, 11u);
    }

    v8f acc[4];
    #pragma unroll
    for (int i = 0; i < 4; i++) acc[i] = (v8f){0.f, 0.f, 0.f, 0.f, 0.f, 0.f, 0.f, 0.f};

    const int mA   = lane & 15;
    const int koff = (lane >> 4) << 3;          // 0 / 8
    const int nB   = lane & 15;
    const int kb   = (lane >> 4) << 3;          // 0 / 8

    for (int s = 0; s < S; s++) {
        __builtin_amdgcn_s_wait_tensorcnt(0);   // wave0: TDM done; others trivially pass
        __syncthreads();                        // publish staged tiles to all waves
        if (wave == 0 && s + 1 < S) {
            // prefetch next A tile into the other buffer while we compute
            tdm_load_tile_f16((unsigned)(size_t)&As[(s + 1) & 1][0],
                              (const void*)(A + (size_t)m0 * K + ((s + 1) << 5)),
                              32u, 128u, (unsigned)K, 3u, 11u);
        }

        const half_t* Ab = As[s & 1];
        h8 a_lo = *(const h8*)(&Ab[(wave * 16 + mA) * 56 + koff]);
        h8 a_hi = *(const h8*)(&Ab[(wave * 16 + mA) * 56 + koff + 16]);
        v16h afrag = __builtin_shufflevector(a_lo, a_hi, 0,1,2,3,4,5,6,7,8,9,10,11,12,13,14,15);

        const int kk = s << 5;
        #pragma unroll
        for (int nt = 0; nt < 4; nt++) {
            h8 b_lo = *(const h8*)(&Bs[(nt * 16 + nB) * bpitch + kk + kb]);
            h8 b_hi = *(const h8*)(&Bs[(nt * 16 + nB) * bpitch + kk + kb + 16]);
            v16h bfrag = __builtin_shufflevector(b_lo, b_hi, 0,1,2,3,4,5,6,7,8,9,10,11,12,13,14,15);
            acc[nt] = __builtin_amdgcn_wmma_f32_16x16x32_f16(
                false, afrag, false, bfrag, (short)0, acc[nt], false, false);
        }
    }

    // Epilogue: bias (+relu), write f32 / f16.
    const int mrb = (lane >> 4) << 3;           // 0 / 8
    const int nl  = lane & 15;
    #pragma unroll
    for (int nt = 0; nt < 4; nt++) {
        int col = n0 + nt * 16 + nl;
        float bv = bias ? bias[col] : 0.0f;
        #pragma unroll
        for (int i = 0; i < 8; i++) {
            int row = m0 + wave * 16 + mrb + i;
            if (row >= Mtot) continue;
            float v = acc[nt][i] + bv;
            if (flags & 4) v = v > 0.f ? v : 0.f;
            size_t o = (size_t)row * Ntot + col;
            if (flags & 1) Cf[o] = v;
            if (flags & 2) Ch[o] = (half_t)v;
        }
    }
}

// ---------------------------------------------------------------------------
// ProbSparse metric: M[b,h,l] = max_u(q.k_s) - sum_u(q.k_s)/L
// qkv layout: [M x 192] fp32, q cols 0..63, k cols 64..127, v cols 128..191
// ---------------------------------------------------------------------------
__global__ __launch_bounds__(256) void metric_kernel(
    const float* __restrict__ qkv, const int* __restrict__ idx, float* __restrict__ Mout)
{
    int t = blockIdx.x * 256 + threadIdx.x;
    if (t >= Bn * Hn * Ln) return;
    int l  = t & (Ln - 1);
    int bh = t >> 13;
    int h  = bh & 7;
    int b  = bh >> 3;
    const float* qr = qkv + (size_t)(b * Ln + l) * 192 + h * En;
    float q[8];
    #pragma unroll
    for (int e = 0; e < 8; e++) q[e] = qr[e];
    float mx = -__builtin_inff(), sum = 0.f;
    for (int u = 0; u < Un; u++) {
        int kl = idx[l * Un + u];
        const float* kr = qkv + (size_t)(b * Ln + kl) * 192 + 64 + h * En;
        float s = 0.f;
        #pragma unroll
        for (int e = 0; e < 8; e++) s += q[e] * kr[e];
        mx = fmaxf(mx, s);
        sum += s;
    }
    Mout[t] = mx - sum * (1.0f / (float)Ln);
}

// ---------------------------------------------------------------------------
// Top-U selection per (b,h): iterative argmax over LDS copy of M row.
// ---------------------------------------------------------------------------
__global__ __launch_bounds__(256) void topk_kernel(
    const float* __restrict__ Mm, int* __restrict__ topk)
{
    __shared__ float sv[Ln];
    __shared__ float rv[256];
    __shared__ int   ri[256];
    int bh = blockIdx.x;
    const float* src = Mm + (size_t)bh * Ln;
    for (int i = threadIdx.x; i < Ln; i += 256) sv[i] = src[i];
    __syncthreads();
    for (int it = 0; it < Un; it++) {
        float best = -__builtin_inff();
        int bi = 0;
        for (int i = threadIdx.x; i < Ln; i += 256) {
            float v = sv[i];
            if (v > best) { best = v; bi = i; }
        }
        rv[threadIdx.x] = best; ri[threadIdx.x] = bi;
        __syncthreads();
        for (int s = 128; s > 0; s >>= 1) {
            if (threadIdx.x < s && rv[threadIdx.x + s] > rv[threadIdx.x]) {
                rv[threadIdx.x] = rv[threadIdx.x + s];
                ri[threadIdx.x] = ri[threadIdx.x + s];
            }
            __syncthreads();
        }
        if (threadIdx.x == 0) {
            topk[(size_t)bh * Un + it] = ri[0];
            sv[ri[0]] = -__builtin_inff();
        }
        __syncthreads();
    }
}

// ---------------------------------------------------------------------------
// Full attention for the U selected queries: two-pass softmax + weighted V.
// One block per (b,h,u).
// ---------------------------------------------------------------------------
__global__ __launch_bounds__(256) void attn_kernel(
    const float* __restrict__ qkv, const int* __restrict__ topk, float* __restrict__ upd)
{
    __shared__ float red[256];
    __shared__ float a8[256][8];
    int g  = blockIdx.x;
    int u  = g % Un;
    int bh = g / Un;
    int h  = bh & 7;
    int b  = bh >> 3;
    int lq = topk[(size_t)bh * Un + u];
    const float* qr = qkv + (size_t)(b * Ln + lq) * 192 + h * En;
    float q[8];
    #pragma unroll
    for (int e = 0; e < 8; e++) q[e] = qr[e] * 0.35355339059f;   // 1/sqrt(E)

    int tid = threadIdx.x;
    float mx = -__builtin_inff();
    for (int l = tid; l < Ln; l += 256) {
        const float* kr = qkv + (size_t)(b * Ln + l) * 192 + 64 + h * En;
        float s = 0.f;
        #pragma unroll
        for (int e = 0; e < 8; e++) s += q[e] * kr[e];
        mx = fmaxf(mx, s);
    }
    red[tid] = mx;
    __syncthreads();
    for (int s = 128; s > 0; s >>= 1) {
        if (tid < s) red[tid] = fmaxf(red[tid], red[tid + s]);
        __syncthreads();
    }
    float gmax = red[0];
    __syncthreads();

    float se = 0.f;
    float acc[8];
    #pragma unroll
    for (int e = 0; e < 8; e++) acc[e] = 0.f;
    for (int l = tid; l < Ln; l += 256) {
        const float* kr = qkv + (size_t)(b * Ln + l) * 192 + 64 + h * En;
        float s = 0.f;
        #pragma unroll
        for (int e = 0; e < 8; e++) s += q[e] * kr[e];
        float p = __expf(s - gmax);
        se += p;
        const float* vr = qkv + (size_t)(b * Ln + l) * 192 + 128 + h * En;
        #pragma unroll
        for (int e = 0; e < 8; e++) acc[e] += p * vr[e];
    }
    red[tid] = se;
    #pragma unroll
    for (int e = 0; e < 8; e++) a8[tid][e] = acc[e];
    __syncthreads();
    for (int s = 128; s > 0; s >>= 1) {
        if (tid < s) {
            red[tid] += red[tid + s];
            #pragma unroll
            for (int e = 0; e < 8; e++) a8[tid][e] += a8[tid + s][e];
        }
        __syncthreads();
    }
    if (tid < 8) upd[(size_t)g * 8 + tid] = a8[0][tid] / red[0];
}

// ---------------------------------------------------------------------------
// v-mean per (b,h), broadcast into ctx, then scatter updated rows.
// ---------------------------------------------------------------------------
__global__ __launch_bounds__(256) void vmean_kernel(const float* __restrict__ qkv, float* __restrict__ vmean)
{
    __shared__ float a8[256][8];
    int bh = blockIdx.x, h = bh & 7, b = bh >> 3, tid = threadIdx.x;
    float acc[8];
    #pragma unroll
    for (int e = 0; e < 8; e++) acc[e] = 0.f;
    for (int l = tid; l < Ln; l += 256) {
        const float* vr = qkv + (size_t)(b * Ln + l) * 192 + 128 + h * En;
        #pragma unroll
        for (int e = 0; e < 8; e++) acc[e] += vr[e];
    }
    #pragma unroll
    for (int e = 0; e < 8; e++) a8[tid][e] = acc[e];
    __syncthreads();
    for (int s = 128; s > 0; s >>= 1) {
        if (tid < s) {
            #pragma unroll
            for (int e = 0; e < 8; e++) a8[tid][e] += a8[tid + s][e];
        }
        __syncthreads();
    }
    if (tid < 8) vmean[(size_t)bh * 8 + tid] = a8[0][tid] * (1.0f / (float)Ln);
}

__global__ __launch_bounds__(256) void ctx_build_kernel(const float* __restrict__ vmean, half_t* __restrict__ ctx)
{
    int t = blockIdx.x * 256 + threadIdx.x;
    if (t >= Mrows * Dm) return;
    int d = t & 63, row = t >> 6, b = row >> 13;
    ctx[t] = (half_t)vmean[((size_t)b * Hn + (d >> 3)) * En + (d & 7)];
}

__global__ __launch_bounds__(256) void scatter_kernel(
    const float* __restrict__ upd, const int* __restrict__ topk, half_t* __restrict__ ctx)
{
    int t = blockIdx.x * 256 + threadIdx.x;
    if (t >= Bn * Hn * Un * En) return;
    int e = t & 7, g = t >> 3;
    int u = g % Un, bh = g / Un, h = bh & 7, b = bh >> 3;
    int l = topk[(size_t)bh * Un + u];
    ctx[(size_t)(b * Ln + l) * 64 + h * En + e] = (half_t)upd[t];
}

// ---------------------------------------------------------------------------
// Fused residual-add + LayerNorm over 64 cols; one wave32 per row (2 cols/lane).
// ---------------------------------------------------------------------------
__global__ __launch_bounds__(256) void ln_kernel(
    const float* __restrict__ xin, const float* __restrict__ res,
    const float* __restrict__ g, const float* __restrict__ bta,
    float* __restrict__ of32, half_t* __restrict__ of16)
{
    int wave = threadIdx.x >> 5, lane = threadIdx.x & 31;
    int row = blockIdx.x * 8 + wave;
    size_t base = (size_t)row * 64;
    float v0 = xin[base + lane], v1 = xin[base + lane + 32];
    if (res) { v0 += res[base + lane]; v1 += res[base + lane + 32]; }
    float s = v0 + v1;
    #pragma unroll
    for (int i = 16; i > 0; i >>= 1) s += __shfl_xor(s, i, 32);
    float mean = s * (1.0f / 64.0f);
    float d0 = v0 - mean, d1 = v1 - mean;
    float vs = d0 * d0 + d1 * d1;
    #pragma unroll
    for (int i = 16; i > 0; i >>= 1) vs += __shfl_xor(vs, i, 32);
    float rstd = rsqrtf(vs * (1.0f / 64.0f) + 1e-5f);
    float o0 = d0 * rstd * g[lane] + bta[lane];
    float o1 = d1 * rstd * g[lane + 32] + bta[lane + 32];
    of32[base + lane] = o0;
    of32[base + lane + 32] = o1;
    if (of16) {
        of16[base + lane] = (half_t)o0;
        of16[base + lane + 32] = (half_t)o1;
    }
}

// ---------------------------------------------------------------------------
// Final projection (64 -> 5) + softmax.
// ---------------------------------------------------------------------------
__global__ __launch_bounds__(256) void proj_softmax_kernel(
    const float* __restrict__ x, const float* __restrict__ pw,
    const float* __restrict__ pb, float* __restrict__ out)
{
    int row = blockIdx.x * 256 + threadIdx.x;
    if (row >= Mrows) return;
    const float* xr = x + (size_t)row * 64;
    float lg[NCn];
    #pragma unroll
    for (int c = 0; c < NCn; c++) lg[c] = pb[c];
    for (int d = 0; d < 64; d++) {
        float xv = xr[d];
        #pragma unroll
        for (int c = 0; c < NCn; c++) lg[c] += xv * pw[d * NCn + c];
    }
    float mx = lg[0];
    #pragma unroll
    for (int c = 1; c < NCn; c++) mx = fmaxf(mx, lg[c]);
    float se = 0.f;
    #pragma unroll
    for (int c = 0; c < NCn; c++) { lg[c] = __expf(lg[c] - mx); se += lg[c]; }
    float inv = 1.0f / se;
    #pragma unroll
    for (int c = 0; c < NCn; c++) out[(size_t)row * NCn + c] = lg[c] * inv;
}

// ---------------------------------------------------------------------------
// Host orchestration
// ---------------------------------------------------------------------------
extern "C" void kernel_launch(void* const* d_in, const int* in_sizes, int n_in,
                              void* d_out, int out_size, void* d_ws, size_t ws_size,
                              hipStream_t stream)
{
    (void)in_sizes; (void)n_in; (void)out_size; (void)ws_size;
    const float* x_enc   = (const float*)d_in[0];
    const int*   idx_all = (const int*)d_in[1];
    const float* emb_w   = (const float*)d_in[2];
    const float* Wq = (const float*)d_in[3];  const float* bq = (const float*)d_in[4];
    const float* Wk = (const float*)d_in[5];  const float* bk = (const float*)d_in[6];
    const float* Wv = (const float*)d_in[7];  const float* bv = (const float*)d_in[8];
    const float* Wo = (const float*)d_in[9];  const float* bo = (const float*)d_in[10];
    const float* c1w = (const float*)d_in[11]; const float* c1b = (const float*)d_in[12];
    const float* c2w = (const float*)d_in[13]; const float* c2b = (const float*)d_in[14];
    const float* ln1g = (const float*)d_in[15]; const float* ln1b = (const float*)d_in[16];
    const float* ln2g = (const float*)d_in[17]; const float* ln2b = (const float*)d_in[18];
    const float* lnfg = (const float*)d_in[19]; const float* lnfb = (const float*)d_in[20];
    const float* pw = (const float*)d_in[21];  const float* pb = (const float*)d_in[22];

    char* p = (char*)d_ws;
    auto carve = [&](size_t bytes) { char* r = p; p += (bytes + 255) & ~(size_t)255; return r; };
    float*  xf32   = (float*) carve((size_t)Mrows * 64 * 4);
    float*  tmp64  = (float*) carve((size_t)Mrows * 64 * 4);
    float*  qkv    = (float*) carve((size_t)Mrows * 192 * 4);
    float*  Mm     = (float*) carve((size_t)Bn * Hn * Ln * 4);
    float*  vmean  = (float*) carve((size_t)Bn * Hn * En * 4);
    float*  upd    = (float*) carve((size_t)Bn * Hn * Un * En * 4);
    int*    topk   = (int*)   carve((size_t)Bn * Hn * Un * 4);
    half_t* xf16   = (half_t*)carve((size_t)Mrows * 64 * 2);
    half_t* ctxf16 = (half_t*)carve((size_t)Mrows * 64 * 2);
    half_t* yf16   = (half_t*)carve((size_t)Mrows * DFFn * 2);
    half_t* wqkvT  = (half_t*)carve((size_t)192 * 64 * 2);
    half_t* woT    = (half_t*)carve((size_t)64 * 64 * 2);
    half_t* w1T    = (half_t*)carve((size_t)DFFn * 64 * 2);
    half_t* w2T    = (half_t*)carve((size_t)64 * DFFn * 2);
    float*  bqkv   = (float*) carve((size_t)192 * 4);

    embed_kernel<<<(Mrows * 64) / 256, 256, 0, stream>>>(x_enc, emb_w, xf32, xf16);

    const int g4096  = (64 * 64 + 255) / 256;
    const int g16384 = (64 * 256 + 255) / 256;

    for (int i = 0; i < NLn; i++) {
        // pack per-layer weights to f16 transposed
        pack_wT_kernel<<<g4096, 256, 0, stream>>>(Wq + (size_t)i * 4096, wqkvT,            64, 64);
        pack_wT_kernel<<<g4096, 256, 0, stream>>>(Wk + (size_t)i * 4096, wqkvT + 64 * 64,  64, 64);
        pack_wT_kernel<<<g4096, 256, 0, stream>>>(Wv + (size_t)i * 4096, wqkvT + 128 * 64, 64, 64);
        pack_wT_kernel<<<g4096, 256, 0, stream>>>(Wo + (size_t)i * 4096, woT, 64, 64);
        pack_wT_kernel<<<g16384, 256, 0, stream>>>(c1w + (size_t)i * 64 * 256, w1T, 64, 256);
        pack_wT_kernel<<<g16384, 256, 0, stream>>>(c2w + (size_t)i * 256 * 64, w2T, 256, 64);
        copy_f32_kernel<<<1, 256, 0, stream>>>(bq + i * 64, bqkv,       64);
        copy_f32_kernel<<<1, 256, 0, stream>>>(bk + i * 64, bqkv + 64,  64);
        copy_f32_kernel<<<1, 256, 0, stream>>>(bv + i * 64, bqkv + 128, 64);

        // fused QKV projection: [32768 x 64] x [64 x 192]
        gemm_wmma_kernel<<<dim3(Mrows / 128, 3), 256, 0, stream>>>(
            xf16, wqkvT, bqkv, qkv, nullptr, Mrows, 192, 64, 1);

        // ProbSparse attention
        metric_kernel<<<(Bn * Hn * Ln) / 256, 256, 0, stream>>>(qkv, idx_all + (size_t)i * Ln * Un, Mm);
        topk_kernel<<<Bn * Hn, 256, 0, stream>>>(Mm, topk);
        attn_kernel<<<Bn * Hn * Un, 256, 0, stream>>>(qkv, topk, upd);
        vmean_kernel<<<Bn * Hn, 256, 0, stream>>>(qkv, vmean);
        ctx_build_kernel<<<(Mrows * 64) / 256, 256, 0, stream>>>(vmean, ctxf16);
        scatter_kernel<<<(Bn * Hn * Un * En + 255) / 256, 256, 0, stream>>>(upd, topk, ctxf16);

        // output projection + residual LN1
        gemm_wmma_kernel<<<dim3(Mrows / 128, 1), 256, 0, stream>>>(
            ctxf16, woT, bo + i * 64, tmp64, nullptr, Mrows, 64, 64, 1);
        ln_kernel<<<Mrows / 8, 256, 0, stream>>>(tmp64, xf32, ln1g + i * 64, ln1b + i * 64, xf32, xf16);

        // FFN: relu(x W1 + b1) W2 + b2, residual LN2
        gemm_wmma_kernel<<<dim3(Mrows / 128, 4), 256, 0, stream>>>(
            xf16, w1T, c1b + i * 256, nullptr, yf16, Mrows, 256, 64, 2 | 4);
        gemm_wmma_kernel<<<dim3(Mrows / 128, 1), 256, 0, stream>>>(
            yf16, w2T, c2b + i * 64, tmp64, nullptr, Mrows, 64, 256, 1);
        ln_kernel<<<Mrows / 8, 256, 0, stream>>>(tmp64, xf32, ln2g + i * 64, ln2b + i * 64, xf32, xf16);
    }

    ln_kernel<<<Mrows / 8, 256, 0, stream>>>(xf32, nullptr, lnfg, lnfb, xf32, nullptr);
    proj_softmax_kernel<<<(Mrows + 255) / 256, 256, 0, stream>>>(xf32, pw, pb, (float*)d_out);
}
